// SelfAttention2d_31482110280100
// MI455X (gfx1250) — matrix operand order repaired and checked by reading, not compiled, not run on hardware
//
#include <hip/hip_runtime.h>
#include <hip/hip_bf16.h>

typedef _Float16 v16h __attribute__((ext_vector_type(16)));
typedef _Float16 v8h  __attribute__((ext_vector_type(8)));
typedef _Float16 v4h  __attribute__((ext_vector_type(4)));
typedef float    v8f  __attribute__((ext_vector_type(8)));

#define C_      512
#define L_      1024
#define NH_     8
#define CH_     64
#define O3C_    1536
#define EPS_    1e-5f
#define QSCALE_ 0.35355339059327373f   // 64^(-0.25) = 2^(-1.5)

#define WMMA(a, b, c) \
  __builtin_amdgcn_wmma_f32_16x16x32_f16(false, (a), false, (b), (short)0, (c), false, false)

// Async 16-byte global -> LDS copy (CDNA5). VDST = LDS byte address (low 32 bits
// of a generic pointer into the LDS aperture ARE the LDS offset, ISA 10.2),
// VADDR = 64-bit global address. Tracked by ASYNCcnt.
static __device__ __forceinline__ void async_cp16(void* lds_dst, const void* gsrc) {
  unsigned loff = (unsigned)(uintptr_t)lds_dst;
  asm volatile("global_load_async_to_lds_b128 %0, %1, off"
               :: "v"(loff), "v"(gsrc) : "memory");
}
static __device__ __forceinline__ void async_wait0() {
  asm volatile("s_wait_asynccnt 0x0" ::: "memory");
}

// A-fragment (16x32 f16, MxK): lane m = lane%16, chunks K=[h*8,h*8+8) and [16+h*8,...)
static __device__ __forceinline__ v16h frag_a_ld(const _Float16* p, int ld) {
  const int lane = threadIdx.x & 31;
  const int m = lane & 15, h = lane >> 4;
  const _Float16* q = p + m * ld + h * 8;
  union { v16h v; v8h h2[2]; } u;
  u.h2[0] = *(const v8h*)q;
  u.h2[1] = *(const v8h*)(q + 16);
  return u.v;
}

// B-fragment (32x16 f16, KxN) from a row-major B^T ([N][K]) tile:
// lane n = lane%16, chunks K=[h*8,h*8+8) and [16+h*8,...)
static __device__ __forceinline__ v16h frag_b_ld(const _Float16* p, int ld) {
  const int lane = threadIdx.x & 31;
  const int n = lane & 15, h = lane >> 4;
  const _Float16* q = p + n * ld + h * 8;
  union { v16h v; v8h h2[2]; } u;
  u.h2[0] = *(const v8h*)q;
  u.h2[1] = *(const v8h*)(q + 16);
  return u.v;
}

// ---------------- Kernel 1: per-sample mean / rstd over C*H*W ----------------
__global__ __launch_bounds__(256) void gn_stats(const float* __restrict__ x,
                                                float* __restrict__ mean,
                                                float* __restrict__ rstd) {
  const int nb = blockIdx.x;
  const size_t cnt = (size_t)C_ * L_;
  const float4* p = (const float4*)(x + (size_t)nb * cnt);
  float s = 0.f, ss = 0.f;
  for (int i = threadIdx.x; i < (int)(cnt / 4); i += 256) {
    float4 v = p[i];
    s  += v.x + v.y + v.z + v.w;
    ss += v.x * v.x + v.y * v.y + v.z * v.z + v.w * v.w;
  }
  __shared__ float sh[512];
  sh[threadIdx.x] = s;
  sh[256 + threadIdx.x] = ss;
  __syncthreads();
  for (int off = 128; off > 0; off >>= 1) {
    if ((int)threadIdx.x < off) {
      sh[threadIdx.x]       += sh[threadIdx.x + off];
      sh[256 + threadIdx.x] += sh[256 + threadIdx.x + off];
    }
    __syncthreads();
  }
  if (threadIdx.x == 0) {
    float m = sh[0] / (float)cnt;
    float var = sh[256] / (float)cnt - m * m;
    mean[nb] = m;
    rstd[nb] = rsqrtf(var + EPS_);
  }
}

// ---------------- Kernel 2: fused norm + QKV GEMM (WMMA) ----------------
__global__ __launch_bounds__(256) void qkv_gemm(
    const float* __restrict__ qkvw, const float* __restrict__ x,
    const float* __restrict__ gw, const float* __restrict__ gb,
    const float* __restrict__ qkvb, const float* __restrict__ mean,
    const float* __restrict__ rstd, _Float16* __restrict__ qkv_ws) {
  const int L0 = blockIdx.x * 128;
  const int M0 = blockIdx.y * 128;
  const int nb = blockIdx.z;
  const int tid = threadIdx.x, wid = tid >> 5, lane = tid & 31;
  const int wm = (wid & 1) * 64, wn = (wid >> 1) * 32;
  const int lh = lane >> 4, nl = lane & 15;

  __shared__ _Float16 As[128 * 40];   // [m][k]
  __shared__ _Float16 Bt[128 * 40];   // [l][k]  (= B^T)

  const float mu = mean[nb], rs = rstd[nb];

  v8f acc[4][2];
#pragma unroll
  for (int i = 0; i < 4; i++)
#pragma unroll
    for (int j = 0; j < 2; j++)
#pragma unroll
      for (int v = 0; v < 8; v++) acc[i][j][v] = 0.f;

  for (int k0 = 0; k0 < C_; k0 += 32) {
    // A tile: float4 loads, packed 4-half LDS stores (k-contiguous)
    for (int t = tid; t < 128 * 8; t += 256) {
      int r = t >> 3, c4 = (t & 7) * 4;
      float4 w4 = *(const float4*)&qkvw[(size_t)(M0 + r) * C_ + (k0 + c4)];
      v4h h;
      h[0] = (_Float16)w4.x; h[1] = (_Float16)w4.y;
      h[2] = (_Float16)w4.z; h[3] = (_Float16)w4.w;
      *(v4h*)&As[r * 40 + c4] = h;
    }
    // B^T tile: fused GroupNorm; float4 loads along l, scalar stores (transpose)
    for (int t = tid; t < 32 * 32; t += 256) {
      int c = t >> 5, l4 = (t & 31) * 4;
      int ch = k0 + c;
      float4 xv = *(const float4*)&x[((size_t)nb * C_ + ch) * L_ + (L0 + l4)];
      float w = rs * gw[ch];
      float b = gb[ch] - mu * w;
      Bt[(l4 + 0) * 40 + c] = (_Float16)(xv.x * w + b);
      Bt[(l4 + 1) * 40 + c] = (_Float16)(xv.y * w + b);
      Bt[(l4 + 2) * 40 + c] = (_Float16)(xv.z * w + b);
      Bt[(l4 + 3) * 40 + c] = (_Float16)(xv.w * w + b);
    }
    __syncthreads();
    v16h af[4], bf[2];
#pragma unroll
    for (int i = 0; i < 4; i++) af[i] = frag_a_ld(&As[(wm + 16 * i) * 40], 40);
#pragma unroll
    for (int j = 0; j < 2; j++) bf[j] = frag_b_ld(&Bt[(wn + 16 * j) * 40], 40);
#pragma unroll
    for (int i = 0; i < 4; i++)
#pragma unroll
      for (int j = 0; j < 2; j++) acc[i][j] = WMMA(af[i], bf[j], acc[i][j]);
    __syncthreads();
  }

#pragma unroll
  for (int i = 0; i < 4; i++) {
    const int row0 = M0 + wm + 16 * i + 8 * lh;  // 8 consecutive output rows
    const int sel  = row0 >> 9;
    const int head = (row0 & 511) >> 6;
    const int ch0  = row0 & 63;
    const float sc = (sel < 2) ? QSCALE_ : 1.f;
#pragma unroll
    for (int j = 0; j < 2; j++) {
      const int l = L0 + wn + 16 * j + nl;
      v8h pk;
#pragma unroll
      for (int v = 0; v < 8; v++)
        pk[v] = (_Float16)((acc[i][j][v] + qkvb[row0 + v]) * sc);
      _Float16* dst = qkv_ws + ((((size_t)sel * 8 + nb) * NH_ + head) * L_ + l) * CH_ + ch0;
      *(v8h*)dst = pk;
    }
  }
}

// ---------------- Kernel 3: flash attention (WMMA + async-to-LDS) ----------------
__global__ __launch_bounds__(256) void flash_attn(const _Float16* __restrict__ qkv_ws,
                                                  _Float16* __restrict__ y_ws) {
  const int qb = blockIdx.x * 128;
  const int head = blockIdx.y;
  const int nb = blockIdx.z;
  const int tid = threadIdx.x, wid = tid >> 5, lane = tid & 31;
  const int lh = lane >> 4, nl = lane & 15;

  const size_t one = (size_t)8 * NH_ * L_ * CH_;
  const _Float16* qp = qkv_ws + ((size_t)nb * NH_ + head) * L_ * CH_;
  const _Float16* kp = qp + one;
  const _Float16* vp = qp + 2 * one;

  __shared__ _Float16 Qs[128 * 72];      // [q_local][ch]
  __shared__ _Float16 Ks[64 * 72];       // [key_local][ch]
  __shared__ _Float16 VTs[64 * 72];      // [ch][key_local]  (V^T)
  __shared__ _Float16 Ps[8 * 16 * 72];   // per-wave P staging [row][key_local]

  // Q tile: async global -> LDS, 16B per lane
  for (int t = tid; t < 128 * 8; t += 256) {
    int r = t >> 3, cc = (t & 7) * 8;
    async_cp16(&Qs[r * 72 + cc], &qp[(size_t)(qb + r) * CH_ + cc]);
  }
  async_wait0();

  v8f Ob[4];
  float mrow[8], lrow[8];
#pragma unroll
  for (int j = 0; j < 4; j++)
#pragma unroll
    for (int v = 0; v < 8; v++) Ob[j][v] = 0.f;
#pragma unroll
  for (int v = 0; v < 8; v++) { mrow[v] = -1e30f; lrow[v] = 0.f; }

  _Float16* pw = &Ps[wid * 16 * 72];

  for (int kt = 0; kt < 16; kt++) {
    const int kbase = kt * 64;
    __syncthreads();
    // K tile: async copy
    for (int t = tid; t < 64 * 8; t += 256) {
      int r = t >> 3, cc = (t & 7) * 8;
      async_cp16(&Ks[r * 72 + cc], &kp[(size_t)(kbase + r) * CH_ + cc]);
    }
    // V^T tile: manual transpose (vector global load, scalar LDS stores)
    for (int t = tid; t < 64 * 8; t += 256) {
      int kl = t >> 3, ch0 = (t & 7) * 8;
      v8h vv = *(const v8h*)&vp[(size_t)(kbase + kl) * CH_ + ch0];
#pragma unroll
      for (int e = 0; e < 8; e++) VTs[(ch0 + e) * 72 + kl] = vv[e];
    }
    async_wait0();
    __syncthreads();

    // S = Q @ K^T  (16 x 64 per wave)
    v16h aq0 = frag_a_ld(&Qs[(wid * 16) * 72 + 0], 72);
    v16h aq1 = frag_a_ld(&Qs[(wid * 16) * 72 + 32], 72);
    v8f S[4];
#pragma unroll
    for (int j = 0; j < 4; j++) {
      v16h b0 = frag_b_ld(&Ks[(16 * j) * 72 + 0], 72);
      v16h b1 = frag_b_ld(&Ks[(16 * j) * 72 + 32], 72);
      v8f s;
#pragma unroll
      for (int v = 0; v < 8; v++) s[v] = 0.f;
      s = WMMA(aq0, b0, s);
      s = WMMA(aq1, b1, s);
      S[j] = s;
    }

    // online softmax: each C-VGPR holds one row across a 16-lane group
#pragma unroll
    for (int v = 0; v < 8; v++) {
      float rm = fmaxf(fmaxf(S[0][v], S[1][v]), fmaxf(S[2][v], S[3][v]));
#pragma unroll
      for (int off = 1; off < 16; off <<= 1) rm = fmaxf(rm, __shfl_xor(rm, off, 32));
      float mn = fmaxf(mrow[v], rm);
      float alpha = __expf(mrow[v] - mn);
      mrow[v] = mn;
      lrow[v] *= alpha;
#pragma unroll
      for (int j = 0; j < 4; j++) Ob[j][v] *= alpha;
    }
#pragma unroll
    for (int j = 0; j < 4; j++)
#pragma unroll
      for (int v = 0; v < 8; v++) S[j][v] = __expf(S[j][v] - mrow[v]);
#pragma unroll
    for (int v = 0; v < 8; v++) {
      float rsum = S[0][v] + S[1][v] + S[2][v] + S[3][v];
#pragma unroll
      for (int off = 1; off < 16; off <<= 1) rsum += __shfl_xor(rsum, off, 32);
      lrow[v] += rsum;
    }

    // stage P (f16) into A-fragment layout (wave-private; LDS ops in-order per wave)
#pragma unroll
    for (int j = 0; j < 4; j++)
#pragma unroll
      for (int v = 0; v < 8; v++)
        pw[(v + 8 * lh) * 72 + 16 * j + nl] = (_Float16)S[j][v];

    // O += P @ V
    v16h p0 = frag_a_ld(pw + 0, 72);
    v16h p1 = frag_a_ld(pw + 32, 72);
#pragma unroll
    for (int j = 0; j < 4; j++) {
      v16h b0 = frag_b_ld(&VTs[(16 * j) * 72 + 0], 72);
      v16h b1 = frag_b_ld(&VTs[(16 * j) * 72 + 32], 72);
      Ob[j] = WMMA(p0, b0, Ob[j]);
      Ob[j] = WMMA(p1, b1, Ob[j]);
    }
  }

  float inv[8];
#pragma unroll
  for (int v = 0; v < 8; v++) inv[v] = 1.f / lrow[v];
#pragma unroll
  for (int j = 0; j < 4; j++)
#pragma unroll
    for (int v = 0; v < 8; v++) {
      int l = qb + wid * 16 + v + 8 * lh;
      int cg = head * CH_ + 16 * j + nl;
      y_ws[((size_t)nb * L_ + l) * C_ + cg] = (_Float16)(Ob[j][v] * inv[v]);
    }
}

// ---------------- Kernel 4: output projection + residual (WMMA) ----------------
__global__ __launch_bounds__(256) void out_gemm(
    const float* __restrict__ ow, const _Float16* __restrict__ y_ws,
    const float* __restrict__ ob, const float* __restrict__ x,
    float* __restrict__ out) {
  const int L0 = blockIdx.x * 128;
  const int M0 = blockIdx.y * 128;
  const int nb = blockIdx.z;
  const int tid = threadIdx.x, wid = tid >> 5, lane = tid & 31;
  const int wm = (wid & 1) * 64, wn = (wid >> 1) * 32;
  const int lh = lane >> 4, nl = lane & 15;

  __shared__ _Float16 As[128 * 40];
  __shared__ _Float16 Bt[128 * 40];

  v8f acc[4][2];
#pragma unroll
  for (int i = 0; i < 4; i++)
#pragma unroll
    for (int j = 0; j < 2; j++)
#pragma unroll
      for (int v = 0; v < 8; v++) acc[i][j][v] = 0.f;

  for (int k0 = 0; k0 < C_; k0 += 32) {
    // A tile: float4 loads + packed 4-half stores
    for (int t = tid; t < 128 * 8; t += 256) {
      int r = t >> 3, c4 = (t & 7) * 4;
      float4 w4 = *(const float4*)&ow[(size_t)(M0 + r) * C_ + (k0 + c4)];
      v4h h;
      h[0] = (_Float16)w4.x; h[1] = (_Float16)w4.y;
      h[2] = (_Float16)w4.z; h[3] = (_Float16)w4.w;
      *(v4h*)&As[r * 40 + c4] = h;
    }
    // B^T tile: y already f16 [l][c] -> async 16B copies
    for (int t = tid; t < 128 * 4; t += 256) {
      int r = t >> 2, cc = (t & 3) * 8;
      async_cp16(&Bt[r * 40 + cc],
                 &y_ws[((size_t)nb * L_ + L0 + r) * C_ + k0 + cc]);
    }
    async_wait0();
    __syncthreads();
    v16h af[4], bf[2];
#pragma unroll
    for (int i = 0; i < 4; i++) af[i] = frag_a_ld(&As[(wm + 16 * i) * 40], 40);
#pragma unroll
    for (int j = 0; j < 2; j++) bf[j] = frag_b_ld(&Bt[(wn + 16 * j) * 40], 40);
#pragma unroll
    for (int i = 0; i < 4; i++)
#pragma unroll
      for (int j = 0; j < 2; j++) acc[i][j] = WMMA(af[i], bf[j], acc[i][j]);
    __syncthreads();
  }

#pragma unroll
  for (int i = 0; i < 4; i++) {
    const int row0 = M0 + wm + 16 * i + 8 * lh;
#pragma unroll
    for (int j = 0; j < 2; j++) {
      const int l = L0 + wn + 16 * j + nl;
#pragma unroll
      for (int v = 0; v < 8; v++) {
        size_t idx = ((size_t)nb * C_ + row0 + v) * L_ + l;
        out[idx] = acc[i][j][v] + ob[row0 + v] + x[idx];
      }
    }
  }
}

extern "C" void kernel_launch(void* const* d_in, const int* in_sizes, int n_in,
                              void* d_out, int out_size, void* d_ws, size_t ws_size,
                              hipStream_t stream) {
  (void)in_sizes; (void)n_in; (void)out_size; (void)ws_size;
  const float* x    = (const float*)d_in[0];
  const float* gw   = (const float*)d_in[1];
  const float* gb   = (const float*)d_in[2];
  const float* qkvw = (const float*)d_in[3];
  const float* qkvb = (const float*)d_in[4];
  const float* ow   = (const float*)d_in[5];
  const float* obv  = (const float*)d_in[6];

  float* mean = (float*)d_ws;
  float* rstd = mean + 8;
  _Float16* qkv_ws = (_Float16*)((char*)d_ws + 256);                  // 3*8*8*1024*64 halfs
  _Float16* y_ws   = qkv_ws + (size_t)3 * 8 * NH_ * L_ * CH_;         // 8*1024*512 halfs

  gn_stats<<<8, 256, 0, stream>>>(x, mean, rstd);
  qkv_gemm<<<dim3(L_ / 128, O3C_ / 128, 8), 256, 0, stream>>>(
      qkvw, x, gw, gb, qkvb, mean, rstd, qkv_ws);
  flash_attn<<<dim3(L_ / 128, NH_, 8), 256, 0, stream>>>(qkv_ws, y_ws);
  out_gemm<<<dim3(L_ / 128, C_ / 128, 8), 256, 0, stream>>>(ow, y_ws, obv, x,
                                                            (float*)d_out);
}
